// CNN_9818295238933
// MI455X (gfx1250) — matrix operand order repaired and checked by reading, not compiled, not run on hardware
//
#include <hip/hip_runtime.h>
#include <hip/hip_bf16.h>

typedef __attribute__((ext_vector_type(16))) _Float16 v16h;
typedef __attribute__((ext_vector_type(8)))  _Float16 v8h;
typedef __attribute__((ext_vector_type(8)))  float    v8f;

#define B_    64
#define CIN   12
#define F_    9000
#define N_    7
#define K_    32
#define FTILES 563          // ceil(9000/16)
#define TPW    4            // f-tiles per wave in gcn stage
#define GRPS   141          // ceil(FTILES/TPW)
#define FC1_IN 288000       // K_*F_
#define SLABS  225
#define SLAB_I 1280
#define CHUNKS 40

// ---------------------------------------------------------------------------
// Stage 1: h1[b,k,f] = sum_{c,n} x[b,c,f,n] * W1[k,c,n]   (f32 VALU, W1 in LDS)
// Output H1 as f16, layout [b][f][k] so each gather row is 64B contiguous.
// ---------------------------------------------------------------------------
__global__ void stage1_kernel(const float* __restrict__ x,
                              const float* __restrict__ W1,
                              _Float16* __restrict__ H1) {
    __shared__ float sw[K_ * CIN * N_];            // 2688 floats
    for (int i = threadIdx.x; i < K_ * CIN * N_; i += blockDim.x) sw[i] = W1[i];
    __syncthreads();

    int gid = blockIdx.x * blockDim.x + threadIdx.x;   // 0 .. B_*F_-1 (exact)
    int b = gid / F_;
    int f = gid - b * F_;

    float acc[K_];
#pragma unroll
    for (int k = 0; k < K_; ++k) acc[k] = 0.f;

    for (int c = 0; c < CIN; ++c) {
        const float* xp = x + (((size_t)b * CIN + c) * F_ + f) * N_;
        float xv[N_];
#pragma unroll
        for (int n = 0; n < N_; ++n) xv[n] = xp[n];
#pragma unroll
        for (int k = 0; k < K_; ++k) {
            const float* wp = &sw[k * CIN * N_ + c * N_];
            float s = 0.f;
#pragma unroll
            for (int n = 0; n < N_; ++n) s += xv[n] * wp[n];
            acc[k] += s;
        }
    }

    _Float16* dst = H1 + (size_t)gid * K_;
#pragma unroll
    for (int g = 0; g < 4; ++g) {
        v8h p;
#pragma unroll
        for (int e = 0; e < 8; ++e) p[e] = (_Float16)acc[g * 8 + e];
        *(v8h*)(dst + g * 8) = p;
    }
}

// ---------------------------------------------------------------------------
// Prep: bake W2/W3 into per-lane B-fragment layout for v_wmma_f32_16x16x32_f16.
// B (32x16, KxN): lane l holds column j=(l&15)+16*jt, K = e<8 ? 8*(l>>4)+e : 16+8*(l>>4)+e-8.
// Fragment index: ((jt*7 + n)*32 + lane)*16 + e
// ---------------------------------------------------------------------------
__global__ void prep_bfrag_kernel(const float* __restrict__ W2,
                                  const float* __restrict__ W3,
                                  _Float16* __restrict__ Bf2,
                                  _Float16* __restrict__ Bf3) {
    int gid = blockIdx.x * blockDim.x + threadIdx.x;
    if (gid >= 2 * 2 * 7 * 32 * 16) return;
    int idx = gid;
    int e    = idx & 15;  idx >>= 4;
    int lane = idx & 31;  idx >>= 5;
    int n    = idx % 7;   idx /= 7;
    int jt   = idx & 1;   idx >>= 1;
    int wsel = idx;
    const float* W = wsel ? W3 : W2;
    _Float16* dst  = wsel ? Bf3 : Bf2;
    int j  = jt * 16 + (lane & 15);
    int Kd = (e < 8) ? (lane >> 4) * 8 + e : 16 + (lane >> 4) * 8 + (e - 8);
    float v = W[((size_t)j * K_ + Kd) * N_ + n];       // W[j][k][n]
    dst[(((size_t)jt * 7 + n) * 32 + lane) * 16 + e] = (_Float16)v;
}

// ---------------------------------------------------------------------------
// Stages 2/3: h[b,j,f] = sum_{k,n} Hin[b, adj[b,f,n], k] * W[j,k,n]
// Per block: async-copy the 14336B B-fragment table global->LDS once
// (global_load_async_to_lds_b128 + s_wait_asynccnt). Each wave processes
// TPW=4 f-tiles (rolled loop, keeps VGPR pressure low -> no scratch spills);
// per tile: 7 K-chunks x 2 j-tiles = 14 WMMAs, B fragments read straight
// from LDS with immediate-offset ds_load_b128 pairs.
// A layout (16-bit 16x32): lane l row m=l&15; elems 0-7 -> k=8*hi+0..7,
// elems 8-15 -> k=16+8*hi+0..7 (hi=l>>4): two 16B gathered loads per lane.
// MODE 0: store f16 [b][f][32]   (feeds next gather)
// MODE 1: store f16 [b][j][F]    (== flat matrix for fc1)
// ---------------------------------------------------------------------------
template <int MODE>
__global__ void gcn_kernel(const _Float16* __restrict__ Hin,
                           const _Float16* __restrict__ Bfrag,
                           const long long* __restrict__ adj,
                           _Float16* __restrict__ Hout) {
    __shared__ __align__(32) _Float16 sB[2 * 7 * 32 * 16];   // 14336 B

    // Async stage of the fragment table into LDS (tracked by ASYNCcnt).
    for (int idx = threadIdx.x; idx < 896; idx += 256) {     // 896 * 16B
        unsigned    lds = (unsigned)(uintptr_t)(&sB[0]) + idx * 16;
        const char* gp  = (const char*)Bfrag + idx * 16;
        asm volatile("global_load_async_to_lds_b128 %0, %1, off"
                     :: "v"(lds), "v"(gp) : "memory");
    }
    asm volatile("s_wait_asynccnt 0x0" ::: "memory");
    __syncthreads();

    int wave = blockIdx.x * (blockDim.x >> 5) + (threadIdx.x >> 5);
    int lane = threadIdx.x & 31;
    int b    = wave / GRPS;
    int grp  = wave - b * GRPS;
    int m    = lane & 15;
    int hi   = lane >> 4;
    const _Float16* sBl = sB + lane * 16;   // per-lane base; chunk q at +q*512

#pragma unroll 1
    for (int t4 = 0; t4 < TPW; ++t4) {
        int tile = grp * TPW + t4;
        if (tile >= FTILES) break;                          // wave-uniform
        int fbase = tile * 16;
        int frow  = fbase + m;
        if (frow >= F_) frow = F_ - 1;                      // clamp last tile
        const long long* ap = adj + ((size_t)b * F_ + frow) * N_;

        v8f c0 = {};
        v8f c1 = {};
#pragma unroll
        for (int n = 0; n < N_; ++n) {
            int fp = (int)ap[n];
            const _Float16* row = Hin + ((size_t)b * F_ + fp) * K_;
            union { v16h v; v8h h[2]; } ua;
            ua.h[0] = *(const v8h*)(row + hi * 8);
            ua.h[1] = *(const v8h*)(row + 16 + hi * 8);
            v16h b0 = *(const v16h*)(sBl + (size_t)n * 512);        // jt=0
            v16h b1 = *(const v16h*)(sBl + (size_t)(7 + n) * 512);  // jt=1
            c0 = __builtin_amdgcn_wmma_f32_16x16x32_f16(false, ua.v, false, b0,
                                                        (short)0, c0, false, false);
            c1 = __builtin_amdgcn_wmma_f32_16x16x32_f16(false, ua.v, false, b1,
                                                        (short)0, c1, false, false);
        }

        // C/D layout: lane l, VGPR r -> (M = r + 8*hi, N = l&15)
        if (MODE == 0) {
#pragma unroll
            for (int r = 0; r < 8; ++r) {
                int f = fbase + 8 * hi + r;
                if (f < F_) {
                    _Float16* o = Hout + ((size_t)b * F_ + f) * K_;
                    o[m]      = (_Float16)c0[r];
                    o[16 + m] = (_Float16)c1[r];
                }
            }
        } else {
            if (fbase + 8 * hi + 8 <= F_) {                 // F_ % 8 == 0
                v8h p0, p1;
#pragma unroll
                for (int r = 0; r < 8; ++r) {
                    p0[r] = (_Float16)c0[r];
                    p1[r] = (_Float16)c1[r];
                }
                size_t f0 = (size_t)fbase + 8 * hi;
                *(v8h*)(Hout + ((size_t)b * K_ + m) * F_ + f0)      = p0;
                *(v8h*)(Hout + ((size_t)b * K_ + 16 + m) * F_ + f0) = p1;
            }
        }
    }
}

// ---------------------------------------------------------------------------
// fc1 split-K: deterministic partials P[s][b*100+o] over slabs of 1280 inputs.
// flat is f16 [b][j*F_+f]; weights f32 (memory-bound: 115 MB streamed once).
// swl is transposed [ii][o] so the hot-loop lane stride is 1 dword
// (conflict-free across the 64 LDS banks).
// ---------------------------------------------------------------------------
__global__ void fc1_partial_kernel(const _Float16* __restrict__ flat,
                                   const float* __restrict__ w,
                                   float* __restrict__ P) {
    __shared__ float sflat[64 * 32];      // [b][ii]
    __shared__ float swl[32 * 100];       // [ii][o]  (transposed)
    int t = threadIdx.x;
    int s = blockIdx.x;

    float acc[25];
    int pb[25], po[25];
#pragma unroll
    for (int r = 0; r < 25; ++r) {
        acc[r] = 0.f;
        int q = t + 256 * r;
        pb[r] = q / 100;
        po[r] = q % 100;
    }

    for (int ch = 0; ch < CHUNKS; ++ch) {
        int i0 = s * SLAB_I + ch * 32;
        // flat: 64x32 f16 loaded as 1024 dword loads, converted to f32 in LDS
#pragma unroll
        for (int r = 0; r < 4; ++r) {
            int idx = t + 256 * r;                 // 0..1023
            int bb = idx >> 4, ii = (idx & 15) * 2;
            union { unsigned u; _Float16 h[2]; } cv;
            cv.u = *(const unsigned*)((const char*)flat +
                                      ((size_t)bb * FC1_IN + i0 + ii) * 2);
            sflat[bb * 32 + ii]     = (float)cv.h[0];
            sflat[bb * 32 + ii + 1] = (float)cv.h[1];
        }
        // weights: 100x32 f32, stored transposed
        for (int idx = t; idx < 3200; idx += 256) {
            int oo = idx >> 5, ii = idx & 31;
            swl[ii * 100 + oo] = w[(size_t)oo * FC1_IN + i0 + ii];
        }
        // prefetch next chunk's weight rows into cache (global_prefetch_b8)
        if (ch + 1 < CHUNKS && t < 100)
            __builtin_prefetch(&w[(size_t)t * FC1_IN + i0 + 32], 0, 0);
        __syncthreads();
#pragma unroll
        for (int r = 0; r < 25; ++r) {
            const float* fp = &sflat[pb[r] * 32];
            float sum = 0.f;
#pragma unroll
            for (int ii = 0; ii < 32; ++ii) sum += fp[ii] * swl[ii * 100 + po[r]];
            acc[r] += sum;
        }
        __syncthreads();
    }
#pragma unroll
    for (int r = 0; r < 25; ++r) P[(size_t)s * 6400 + t + 256 * r] = acc[r];
}

// Reduce partials, add bias, batch-norm over B, ReLU. Block per o, thread per b.
__global__ void fc1_reduce_bn_kernel(const float* __restrict__ P,
                                     const float* __restrict__ bias,
                                     const float* __restrict__ gamma,
                                     const float* __restrict__ beta,
                                     float* __restrict__ Y1) {
    int o = blockIdx.x;
    int b = threadIdx.x;
    float v = bias[o];
    for (int s = 0; s < SLABS; ++s) v += P[(size_t)s * 6400 + b * 100 + o];
    __shared__ float sh[64];
    sh[b] = v;
    __syncthreads();
    float m = 0.f;
    for (int j = 0; j < 64; ++j) m += sh[j];
    m *= (1.f / 64.f);
    float var = 0.f;
    for (int j = 0; j < 64; ++j) { float d = sh[j] - m; var += d * d; }
    var *= (1.f / 64.f);
    float y = (v - m) * rsqrtf(var + 1e-5f) * gamma[o] + beta[o];
    Y1[b * 100 + o] = fmaxf(y, 0.f);
}

// fc2 + bn2 + relu + fco, single block.
__global__ void head_kernel(const float* __restrict__ Y1,
                            const float* __restrict__ fc2_w,
                            const float* __restrict__ fc2_b,
                            const float* __restrict__ g2,
                            const float* __restrict__ b2,
                            const float* __restrict__ fco_w,
                            const float* __restrict__ fco_b,
                            float* __restrict__ out) {
    __shared__ float sy[64 * 100];
    __shared__ float sz[64 * 30];
    __shared__ float scale[30], shift[30];
    int t = threadIdx.x;
    for (int i = t; i < 6400; i += 256) sy[i] = Y1[i];
    __syncthreads();
    for (int i = t; i < 1920; i += 256) {
        int b = i / 30, o = i % 30;
        float acc = fc2_b[o];
        for (int j = 0; j < 100; ++j) acc += sy[b * 100 + j] * fc2_w[o * 100 + j];
        sz[i] = acc;
    }
    __syncthreads();
    if (t < 30) {
        float m = 0.f;
        for (int b = 0; b < 64; ++b) m += sz[b * 30 + t];
        m *= (1.f / 64.f);
        float v = 0.f;
        for (int b = 0; b < 64; ++b) { float d = sz[b * 30 + t] - m; v += d * d; }
        v *= (1.f / 64.f);
        float sc = rsqrtf(v + 1e-5f) * g2[t];
        scale[t] = sc;
        shift[t] = b2[t] - m * sc;
    }
    __syncthreads();
    if (t < 128) {
        int b = t >> 1, cls = t & 1;
        float acc = fco_b[cls];
        for (int o = 0; o < 30; ++o) {
            float y = fmaxf(sz[b * 30 + o] * scale[o] + shift[o], 0.f);
            acc += y * fco_w[cls * 30 + o];
        }
        out[b * 2 + cls] = acc;
    }
}

// ---------------------------------------------------------------------------
extern "C" void kernel_launch(void* const* d_in, const int* in_sizes, int n_in,
                              void* d_out, int out_size, void* d_ws, size_t ws_size,
                              hipStream_t stream) {
    (void)in_sizes; (void)n_in; (void)out_size; (void)ws_size;
    const float*     x     = (const float*)d_in[0];
    const long long* adj   = (const long long*)d_in[1];   // int64 indices
    const float*     W1    = (const float*)d_in[2];
    const float*     W2    = (const float*)d_in[3];
    const float*     W3    = (const float*)d_in[4];
    const float*     fc1_w = (const float*)d_in[5];
    const float*     fc1_b = (const float*)d_in[6];
    const float*     bn1_g = (const float*)d_in[7];
    const float*     bn1_b = (const float*)d_in[8];
    const float*     fc2_w = (const float*)d_in[9];
    const float*     fc2_b = (const float*)d_in[10];
    const float*     bn2_g = (const float*)d_in[11];
    const float*     bn2_b = (const float*)d_in[12];
    const float*     fco_w = (const float*)d_in[13];
    const float*     fco_b = (const float*)d_in[14];

    char* ws = (char*)d_ws;
    const size_t H_BYTES = (size_t)B_ * F_ * K_ * 2;     // 36,864,000
    _Float16* H1  = (_Float16*)(ws);                     // stage1 out; later H3/flat
    _Float16* H2  = (_Float16*)(ws + H_BYTES);
    _Float16* Bf2 = (_Float16*)(ws + 2 * H_BYTES);
    _Float16* Bf3 = (_Float16*)(ws + 2 * H_BYTES + 14336);
    float*    P   = (float*)(ws + 2 * H_BYTES + 2 * 14336);
    float*    Y1  = (float*)(ws + 2 * H_BYTES + 2 * 14336 + (size_t)SLABS * 6400 * 4);
    float*    out = (float*)d_out;

    prep_bfrag_kernel<<<(14336 + 255) / 256, 256, 0, stream>>>(W2, W3, Bf2, Bf3);
    stage1_kernel<<<(B_ * F_) / 256, 256, 0, stream>>>(x, W1, H1);
    gcn_kernel<0><<<(B_ * GRPS) / 8, 256, 0, stream>>>(H1, Bf2, adj, H2);
    gcn_kernel<1><<<(B_ * GRPS) / 8, 256, 0, stream>>>(H2, Bf3, adj, H1); // H3 overlays H1
    fc1_partial_kernel<<<SLABS, 256, 0, stream>>>(H1, fc1_w, P);
    fc1_reduce_bn_kernel<<<100, 64, 0, stream>>>(P, fc1_b, bn1_g, bn1_b, Y1);
    head_kernel<<<1, 256, 0, stream>>>(Y1, fc2_w, fc2_b, bn2_g, bn2_b, fco_w, fco_b, out);
}
